// DeformBottleneckBlock_17446157156492
// MI455X (gfx1250) — matrix operand order repaired and checked by reading, not compiled, not run on hardware
//
#include <hip/hip_runtime.h>
#include <hip/hip_bf16.h>

// ---------------- problem constants ----------------
#define BN_   4
#define CIN_  1024
#define H_    100
#define W_    100
#define P_    (H_ * W_)       // 10000 pixels per batch image
#define CB_   256
#define COUT_ 1024
#define BP_   (BN_ * P_)      // 40000 (batch folded into GEMM M; 40000 % 64 == 0)
#define KCAT_ (9 * CB_)       // 2304: fused deformable K (9 taps x 256 ch)

// ---------------- types ----------------
typedef __bf16 bf16_t;
typedef __attribute__((ext_vector_type(16))) __bf16 v16bf;
typedef __attribute__((ext_vector_type(8)))  __bf16 v8bf;
typedef __attribute__((ext_vector_type(8)))  float  v8f;

union frag16 {
  v16bf v;
  struct { v8bf lo, hi; } s;
};

static __device__ __forceinline__ bf16_t f2bf(float f) {
  unsigned u = __builtin_bit_cast(unsigned, f);
  u += 0x7FFFu + ((u >> 16) & 1u);              // round-to-nearest-even
  unsigned short h = (unsigned short)(u >> 16);
  return __builtin_bit_cast(bf16_t, h);
}

// =====================================================================
// bf16 WMMA GEMM:  D[m, n] = sum_k A[m, k] * Bw[n, k]
//   A  : activations, pixel-major [Mtotal, K] bf16 (row-contiguous b128s)
//   Bw : weights [Ntotal, K] bf16 (row-contiguous, no transpose needed)
// Wave tile: 32 M x 64 N  (2 A-frags x 4 B-frags -> 8 WMMAs / K-step).
// Block: 8 waves as 2 M-waves x 4 N-waves -> 64 M x 256 N.
// K-loop is software-pipelined: next-step fragments are loaded into
// registers before the current step's WMMAs issue (unroll 2 => ping-pong
// buffers via renaming, loads get a full step of latency hiding).
// Epilogues:
//   EPI=0: f32  Out[m*N+n] = relu(d*scale[n]+bias[n])        (conv1)
//   EPI=1: bf16 Out[m*N+n] = relu(d*scale[n]+bias[n])        (fused taps+BN2)
//   EPI=2: f32  Out[NCHW]  = relu(d*scale[n]+bias[n]+resid)  (conv3+residual)
// =====================================================================
template <int EPI>
__global__ __launch_bounds__(256) void gemm_bf16_wmma(
    const bf16_t* __restrict__ A,
    const bf16_t* __restrict__ Bw,
    int K, int Ntotal,
    void* __restrict__ OutV,
    const float* __restrict__ scale,
    const float* __restrict__ bias,
    const float* __restrict__ resid,
    int Pp, int Cout)
{
  const int wave = threadIdx.x >> 5;
  const int lane = threadIdx.x & 31;
  const int hlf  = lane >> 4;      // half-wave: K sub-range select
  const int ridx = lane & 15;      // A: row-in-tile (M), B: col-in-tile (N)
  const int wm   = wave >> 2;      // 0..1  (M sub-block)
  const int wn   = wave & 3;       // 0..3  (N sub-block)

  const int m0 = (blockIdx.x * 2 + wm) * 32;
  const int n0 = blockIdx.y * 256 + wn * 64;

  v8f acc[8];
#pragma unroll
  for (int i = 0; i < 8; ++i) acc[i] = v8f{};

  // A fragment (16x32): lane l<16 holds M=l, K={0..7,16..23}; l+16 -> {8..15,24..31}
  const bf16_t* arow0 = A + (size_t)(m0 + ridx) * K + hlf * 8;
  const bf16_t* arow1 = arow0 + (size_t)16 * K;
  // B fragment (32x16): lane l<16 holds N=l, K={0..7,16..23}; l+16 -> {8..15,24..31}
  const bf16_t* brow[4];
#pragma unroll
  for (int j = 0; j < 4; ++j)
    brow[j] = Bw + (size_t)(n0 + j * 16 + ridx) * K + hlf * 8;

  frag16 a[2], b[4];
  a[0].s.lo = *(const v8bf*)(arow0);
  a[0].s.hi = *(const v8bf*)(arow0 + 16);
  a[1].s.lo = *(const v8bf*)(arow1);
  a[1].s.hi = *(const v8bf*)(arow1 + 16);
#pragma unroll
  for (int j = 0; j < 4; ++j) {
    b[j].s.lo = *(const v8bf*)(brow[j]);
    b[j].s.hi = *(const v8bf*)(brow[j] + 16);
  }

#pragma unroll 2
  for (int kb = 32; kb < K; kb += 32) {
    frag16 na[2], nb[4];
    na[0].s.lo = *(const v8bf*)(arow0 + kb);
    na[0].s.hi = *(const v8bf*)(arow0 + kb + 16);
    na[1].s.lo = *(const v8bf*)(arow1 + kb);
    na[1].s.hi = *(const v8bf*)(arow1 + kb + 16);
#pragma unroll
    for (int j = 0; j < 4; ++j) {
      nb[j].s.lo = *(const v8bf*)(brow[j] + kb);
      nb[j].s.hi = *(const v8bf*)(brow[j] + kb + 16);
    }
    if (kb + 64 < K) __builtin_prefetch(arow0 + kb + 64, 0, 1);
#pragma unroll
    for (int i = 0; i < 2; ++i)
#pragma unroll
      for (int j = 0; j < 4; ++j)
        acc[i * 4 + j] = __builtin_amdgcn_wmma_f32_16x16x32_bf16(
            false, a[i].v, false, b[j].v, (short)0, acc[i * 4 + j], false, false);
#pragma unroll
    for (int i = 0; i < 2; ++i) a[i] = na[i];
#pragma unroll
    for (int j = 0; j < 4; ++j) b[j] = nb[j];
  }
  // drain: last K-step
#pragma unroll
  for (int i = 0; i < 2; ++i)
#pragma unroll
    for (int j = 0; j < 4; ++j)
      acc[i * 4 + j] = __builtin_amdgcn_wmma_f32_16x16x32_bf16(
          false, a[i].v, false, b[j].v, (short)0, acc[i * 4 + j], false, false);

  // C/D layout: VGPR r -> M = base + r + hlf*8, N = subtile + ridx
#pragma unroll
  for (int i = 0; i < 2; ++i) {
#pragma unroll
    for (int j = 0; j < 4; ++j) {
      const int n = n0 + j * 16 + ridx;
#pragma unroll
      for (int r = 0; r < 8; ++r) {
        const int m = m0 + i * 16 + r + hlf * 8;
        const float d = acc[i * 4 + j][r];
        if (EPI == 0) {
          ((float*)OutV)[(size_t)m * Ntotal + n] =
              fmaxf(d * scale[n] + bias[n], 0.0f);
        } else if (EPI == 1) {
          ((bf16_t*)OutV)[(size_t)m * Ntotal + n] =
              f2bf(fmaxf(d * scale[n] + bias[n], 0.0f));
        } else {
          const int bb = m / Pp, p = m - bb * Pp;
          const size_t o = ((size_t)bb * Cout + n) * Pp + p;
          ((float*)OutV)[o] = fmaxf(d * scale[n] + bias[n] + resid[o], 0.0f);
        }
      }
    }
  }
}

// =====================================================================
// x [B,CIN,P] f32 (NCHW)  ->  x^T [B*P, CIN] bf16, LDS 32x33 tile transpose
// =====================================================================
__global__ __launch_bounds__(256) void k_transpose_x(
    const float* __restrict__ x, bf16_t* __restrict__ xt)
{
  __shared__ float tile[32][33];
  const int b  = blockIdx.z;
  const int c0 = blockIdx.y * 32;
  const int p0 = blockIdx.x * 32;
  const int tx = threadIdx.x & 31;
  const int ty = threadIdx.x >> 5;  // 0..7
  const float* xb = x + (size_t)b * CIN_ * P_;
#pragma unroll
  for (int i = 0; i < 32; i += 8) {
    const int p = p0 + tx;
    tile[ty + i][tx] = (p < P_) ? xb[(size_t)(c0 + ty + i) * P_ + p] : 0.0f;
  }
  __syncthreads();
  bf16_t* xtb = xt + (size_t)b * P_ * CIN_;
#pragma unroll
  for (int i = 0; i < 32; i += 8) {
    const int p = p0 + ty + i;
    if (p < P_) xtb[(size_t)p * CIN_ + (c0 + tx)] = f2bf(tile[tx][ty + i]);
  }
}

// =====================================================================
// Offset conv (3x3, 256->27, pad 1): block per (b, pixel), 256 threads = c.
// Activation reads are contiguous 1KB rows (pixel-major out1).
// =====================================================================
__global__ __launch_bounds__(256) void k_offset_conv(
    const float* __restrict__ out1,   // [B*P, CB] pixel-major
    const float* __restrict__ wp,     // [27][9][CB] packed weights
    const float* __restrict__ boff,   // [27]
    float* __restrict__ om)           // [B][27][P]
{
  const int b = blockIdx.y;
  const int p = blockIdx.x;
  const int c = threadIdx.x;
  const int h = p / W_, w = p - (p / W_) * W_;

  float a[9];
#pragma unroll
  for (int t = 0; t < 9; ++t) {
    const int ti = t / 3 - 1, tj = t % 3 - 1;
    const int hh = h + ti, ww = w + tj;
    a[t] = (hh >= 0 && hh < H_ && ww >= 0 && ww < W_)
         ? out1[((size_t)b * P_ + (hh * W_ + ww)) * CB_ + c] : 0.0f;
  }

  __shared__ float red[8];
  for (int ko = 0; ko < 27; ++ko) {
    float partial = 0.0f;
#pragma unroll
    for (int t = 0; t < 9; ++t)
      partial += a[t] * wp[((size_t)ko * 9 + t) * CB_ + c];
    for (int o = 16; o > 0; o >>= 1) partial += __shfl_xor(partial, o, 32);
    if ((threadIdx.x & 31) == 0) red[threadIdx.x >> 5] = partial;
    __syncthreads();
    if (threadIdx.x == 0) {
      float s = boff[ko];
#pragma unroll
      for (int i = 0; i < 8; ++i) s += red[i];
      om[((size_t)b * 27 + ko) * P_ + p] = s;
    }
    __syncthreads();
  }
}

// =====================================================================
// Deformable bilinear sampling for tap k -> column slice of the fused
// K=2304 activation matrix: s_all[(b*P+p)*KCAT + k*256 + c].
// dy = om[2k], dx = om[2k+1], mask = sigmoid(om[18+k])  (per reference).
// Each corner read is a contiguous 256-channel row -> coalesced.
// =====================================================================
__global__ __launch_bounds__(256) void k_sample(
    const float* __restrict__ out1,   // [B*P, CB] pixel-major f32
    const float* __restrict__ om,     // [B][27][P]
    bf16_t* __restrict__ s_all,       // [B*P, KCAT] bf16
    int k)
{
  const int b = blockIdx.y;
  const int p = blockIdx.x;
  const int c = threadIdx.x;
  const int h = p / W_, w = p - (p / W_) * W_;
  const int ki = k / 3, kj = k - ki * 3;

  const float* omb = om + (size_t)b * 27 * P_;
  const float dy = omb[(size_t)(2 * k) * P_ + p];
  const float dx = omb[(size_t)(2 * k + 1) * P_ + p];
  const float mz = omb[(size_t)(18 + k) * P_ + p];
  const float mk = 1.0f / (1.0f + __expf(-mz));

  const float yy = (float)(h + ki - 1) + dy;
  const float xx = (float)(w + kj - 1) + dx;
  const float y0f = floorf(yy), x0f = floorf(xx);
  const int y0 = (int)y0f, x0 = (int)x0f;
  const float wy = yy - y0f, wx = xx - x0f;

  const float* base = out1 + (size_t)b * P_ * CB_;
  auto g = [&](int yi, int xi) -> float {
    if (yi < 0 || yi >= H_ || xi < 0 || xi >= W_) return 0.0f;
    return base[((size_t)(yi * W_ + xi)) * CB_ + c];
  };
  const float v = g(y0, x0)         * (1.0f - wy) * (1.0f - wx)
                + g(y0, x0 + 1)     * (1.0f - wy) * wx
                + g(y0 + 1, x0)     * wy          * (1.0f - wx)
                + g(y0 + 1, x0 + 1) * wy          * wx;

  s_all[((size_t)b * P_ + p) * KCAT_ + k * CB_ + c] = f2bf(v * mk);
}

// ---------------- small conversion kernels ----------------
__global__ void k_cvt(const float* __restrict__ in, bf16_t* __restrict__ out, int n) {
  const int i = blockIdx.x * blockDim.x + threadIdx.x;
  if (i < n) out[i] = f2bf(in[i]);
}

// w2 [O=256][C=256][3][3] f32 -> w2cat [O][KCAT=2304] bf16 (tap-major K,
// matching s_all's K layout: K index = t*256 + c)
__global__ void k_cvt_w2cat(const float* __restrict__ w2, bf16_t* __restrict__ o) {
  const int i = blockIdx.x * blockDim.x + threadIdx.x;
  if (i >= CB_ * KCAT_) return;
  const int oo = i / KCAT_;
  const int r  = i - oo * KCAT_;
  const int t  = r >> 8;     // tap
  const int c  = r & 255;    // channel
  o[i] = f2bf(w2[((size_t)oo * CB_ + c) * 9 + t]);
}

// w_off [27][C=256][3][3] f32 -> wp [27][9][C] f32 (c-contiguous)
__global__ void k_pack_woff(const float* __restrict__ w, float* __restrict__ o) {
  const int i = blockIdx.x * blockDim.x + threadIdx.x;
  if (i >= 27 * 9 * CB_) return;
  const int ko = i / (9 * CB_);
  const int r  = i - ko * (9 * CB_);
  const int t  = r / CB_;
  const int c  = r & 255;
  o[i] = w[((size_t)ko * CB_ + c) * 9 + t];
}

// =====================================================================
extern "C" void kernel_launch(void* const* d_in, const int* in_sizes, int n_in,
                              void* d_out, int out_size, void* d_ws, size_t ws_size,
                              hipStream_t stream)
{
  (void)in_sizes; (void)n_in; (void)out_size; (void)ws_size;

  const float* x    = (const float*)d_in[0];
  const float* w1   = (const float*)d_in[1];
  const float* s1   = (const float*)d_in[2];
  const float* b1   = (const float*)d_in[3];
  const float* woff = (const float*)d_in[4];
  const float* boff = (const float*)d_in[5];
  const float* w2   = (const float*)d_in[6];
  const float* s2   = (const float*)d_in[7];
  const float* b2   = (const float*)d_in[8];
  const float* w3   = (const float*)d_in[9];
  const float* s3   = (const float*)d_in[10];
  const float* b3   = (const float*)d_in[11];

  // ---------------- workspace layout (~252 MB peak) ----------------
  char* ws = (char*)d_ws;
  size_t off = 0;
  auto take = [&](size_t bytes) -> char* {
    char* p = ws + off;
    off += (bytes + 255) & ~(size_t)255;
    return p;
  };
  // s_all [40000, 2304] bf16 = 184.3 MB. x^T (81.9 MB) ALIASES its head:
  // x^T is dead after conv1, and sampling (which overwrites the region)
  // is stream-ordered after conv1.
  bf16_t* s_all = (bf16_t*)take((size_t)BP_ * KCAT_ * 2);
  bf16_t* xt    = (bf16_t*)s_all;
  float*  out1  = (float*) take((size_t)BP_ * CB_ * 4);      // 41.0 MB
  float*  om    = (float*) take((size_t)BN_ * 27 * P_ * 4);  // 4.3 MB
  bf16_t* o2b   = (bf16_t*)take((size_t)BP_ * CB_ * 2);      // 20.5 MB
  bf16_t* w1b   = (bf16_t*)take((size_t)CB_ * CIN_ * 2);
  bf16_t* w2cat = (bf16_t*)take((size_t)CB_ * KCAT_ * 2);
  bf16_t* w3b   = (bf16_t*)take((size_t)COUT_ * CB_ * 2);
  float*  wp    = (float*) take((size_t)27 * 9 * CB_ * 4);

  // 1) weight conversion / packing
  k_cvt<<<(CB_ * CIN_ + 255) / 256, 256, 0, stream>>>(w1, w1b, CB_ * CIN_);
  k_cvt<<<(COUT_ * CB_ + 255) / 256, 256, 0, stream>>>(w3, w3b, COUT_ * CB_);
  k_cvt_w2cat<<<(CB_ * KCAT_ + 255) / 256, 256, 0, stream>>>(w2, w2cat);
  k_pack_woff<<<(27 * 9 * CB_ + 255) / 256, 256, 0, stream>>>(woff, wp);

  // 2) x -> pixel-major bf16 (x^T fits the 192 MB L2 for conv1 reuse)
  k_transpose_x<<<dim3((P_ + 31) / 32, CIN_ / 32, BN_), 256, 0, stream>>>(x, xt);

  const int gmx = BP_ / 64;  // 625 blocks, exact (no tail handling)

  // 3) conv1 (1024->256) + BN1 + ReLU   [K=1024, N=256 -> grid.y=1]
  gemm_bf16_wmma<0><<<dim3(gmx, 1), 256, 0, stream>>>(
      xt, w1b, CIN_, CB_, out1, s1, b1, nullptr, 0, 0);

  // 4) 3x3 offset conv -> om (27 channels)
  k_offset_conv<<<dim3(P_, BN_), 256, 0, stream>>>(out1, wp, boff, om);

  // 5) bilinear sample + mask all 9 taps into the fused K=2304 matrix
  for (int k = 0; k < 9; ++k)
    k_sample<<<dim3(P_, BN_), 256, 0, stream>>>(out1, om, s_all, k);

  // 6) ONE fused deformable GEMM (K=2304) + BN2 + ReLU -> bf16
  //    (replaces 9 accumulate GEMMs: kills ~740 MB of f32 acc RMW traffic)
  gemm_bf16_wmma<1><<<dim3(gmx, 1), 256, 0, stream>>>(
      s_all, w2cat, KCAT_, CB_, o2b, s2, b2, nullptr, 0, 0);

  // 7) conv3 (256->1024) + BN3 + residual(x) + ReLU -> d_out (NCHW)
  gemm_bf16_wmma<2><<<dim3(gmx, COUT_ / 256), 256, 0, stream>>>(
      o2b, w3b, CB_, COUT_, (float*)d_out, s3, b3, x, P_, COUT_);
}
